// GraphAttention_5574867550374
// MI455X (gfx1250) — compile-verified, matrix operand order checked by reading, not run on hardware
//
#include <hip/hip_runtime.h>
#include <stdint.h>

#define NN 4096
#define FF 256
#define UU 128
#define HH 8
#define LEAKY 0.2f

typedef __attribute__((ext_vector_type(16))) __bf16 v16bf;
typedef __attribute__((ext_vector_type(8)))  float  v8f;

union BF16Frag { v16bf v; uint32_t u[8]; };

__device__ __forceinline__ uint16_t f2bf_bits(float x) {
    uint32_t u = __builtin_bit_cast(uint32_t, x);
    u += 0x7FFFu + ((u >> 16) & 1u);          // round-to-nearest-even
    return (uint16_t)(u >> 16);
}

__device__ __forceinline__ uint32_t pack_bf2(float lo, float hi) {
    return (uint32_t)f2bf_bits(lo) | ((uint32_t)f2bf_bits(hi) << 16);
}

// 16-bit A-matrix 16x32 layout: VGPR v holds K pair starting at kbase(v,half)
__device__ __forceinline__ int kbase(int v, int half) {
    return ((v < 4) ? (2 * v) : (16 + 2 * (v - 4))) + half * 8;
}

// ---------------- Kernel 0: pack adjacency to bitmask (64MB -> 2MB) --------
__global__ __launch_bounds__(256) void pack_adj_kernel(const int* __restrict__ A,
                                                       uint32_t* __restrict__ Apack) {
    int lane = threadIdx.x & 31;
    int wave = threadIdx.x >> 5;
    int base = blockIdx.x * 256 + wave * 32;          // 32 consecutive A entries
    int a = A[base + lane];
    unsigned long long b = __ballot(a != 0);          // wave32: low 32 bits valid
    if (lane == 0) Apack[base >> 5] = (uint32_t)b;
}

// ---------------- Kernel 0b: W[H,F,U] -> bf16 Wt[H,U,F] --------------------
__global__ __launch_bounds__(256) void transpose_w_kernel(const float* __restrict__ K,
                                                          uint16_t* __restrict__ Wt) {
    int id = blockIdx.x * 256 + threadIdx.x;          // H*U*F
    int h = id / (UU * FF);
    int rem = id - h * (UU * FF);
    int u = rem / FF, f = rem - (rem / FF) * FF;
    Wt[(h * UU + u) * FF + f] = f2bf_bits(K[(h * FF + f) * UU + u]);
}

// ---------------- Kernel 0c: X fp32 -> packed bf16 pairs -------------------
__global__ __launch_bounds__(256) void convert_x_kernel(const float* __restrict__ X,
                                                        uint32_t* __restrict__ Xbf) {
    int id = blockIdx.x * 256 + threadIdx.x;          // N*F/2 packed pairs
    float2 x2 = *(const float2*)(X + id * 2);
    Xbf[id] = pack_bf2(x2.x, x2.y);
}

// ---------------- Kernel 1: feats = X @ W_h, WMMA bf16 ---------------------
__global__ __launch_bounds__(256) void feats_kernel(const uint32_t* __restrict__ Xbf,
                                                    const uint32_t* __restrict__ Wt32,
                                                    float* __restrict__ feats,
                                                    uint16_t* __restrict__ featsT) {
    int lane = threadIdx.x & 31, wave = threadIdx.x >> 5;
    int wid = blockIdx.x * 8 + wave;                  // 0..16383
    int h  = wid >> 11;                               // 2048 waves per head
    int rem = wid & 2047;
    int nt = rem >> 3, ut = rem & 7;
    int half = lane >> 4, l15 = lane & 15;
    int row  = nt * 16 + l15;
    int ucol = ut * 16 + l15;

    const uint32_t* xrow = Xbf  + ((row * FF) >> 1);
    const uint32_t* wrow = Wt32 + (((h * UU + ucol) * FF) >> 1);

    v8f acc;
    #pragma unroll
    for (int j = 0; j < 8; ++j) acc[j] = 0.f;

    for (int kt = 0; kt < 8; ++kt) {                  // K = 256 in steps of 32
        BF16Frag a, b;
        #pragma unroll
        for (int v = 0; v < 8; ++v) {
            int k0 = kt * 32 + kbase(v, half);        // even -> packed pair idx
            a.u[v] = xrow[k0 >> 1];
            b.u[v] = wrow[k0 >> 1];
        }
        acc = __builtin_amdgcn_wmma_f32_16x16x32_bf16(false, a.v, false, b.v,
                                                      (short)0, acc, false, false);
    }
    #pragma unroll
    for (int r = 0; r < 8; ++r) {                     // C layout: M = r + 8*half
        int n = nt * 16 + r + half * 8;
        feats[(h * NN + n) * UU + ucol] = acc[r];
        featsT[(h * UU + ucol) * NN + n] = f2bf_bits(acc[r]);
    }
}

// ---------------- Kernel 2: a_s / a_n projections --------------------------
__global__ __launch_bounds__(256) void attn_proj_kernel(const float* __restrict__ feats,
                                                        const float* __restrict__ attn_self,
                                                        const float* __restrict__ attn_neigh,
                                                        float* __restrict__ a_s,
                                                        float* __restrict__ a_n) {
    int id = blockIdx.x * 256 + threadIdx.x;          // H*N
    int h = id >> 12, n = id & (NN - 1);
    const float* fp = feats + (h * NN + n) * UU;
    const float* ws = attn_self  + h * UU;
    const float* wn = attn_neigh + h * UU;
    float ss = 0.f, sn = 0.f;
    #pragma unroll 4
    for (int u = 0; u < UU; ++u) { float f = fp[u]; ss += f * ws[u]; sn += f * wn[u]; }
    a_s[id] = ss; a_n[id] = sn;
}

// ------- Kernel 3: fused masked softmax + P@V + ELU + head-mean ------------
__global__ __launch_bounds__(256) void gat_kernel(const uint32_t* __restrict__ Apack,
                                                  const uint32_t* __restrict__ ft32,
                                                  const float* __restrict__ a_s,
                                                  const float* __restrict__ a_n,
                                                  const float* __restrict__ biases,
                                                  float* __restrict__ out) {
    __shared__ float red[16 * UU];
    int tid = threadIdx.x;
    for (int i = tid; i < 16 * UU; i += 256) red[i] = 0.f;
    __syncthreads();

    int lane = tid & 31;
    int h = tid >> 5;                                 // one wave per head
    int rb = blockIdx.x;                              // 16-row block
    int half = lane >> 4, l15 = lane & 15;
    int n = rb * 16 + l15;

    float as_v = a_s[h * NN + n];
    const float* anh = a_n + h * NN;
    const uint32_t* arow = Apack + n * (NN / 32);

    // Pass 1: rowmax = leaky(a_s + max_{edge} a_n)  (leaky is monotone)
    float cmax = -3.0e38f;
    int wbase = half * 64;                            // halves split the scan
    for (int w = 0; w < 64; ++w) {
        uint32_t word = arow[wbase + w];
        const float4* p4 = (const float4*)(anh + (wbase + w) * 32);
        #pragma unroll
        for (int q = 0; q < 8; ++q) {
            float4 v4 = p4[q];
            int b = q * 4;
            cmax = ((word >> (b + 0)) & 1u) ? fmaxf(cmax, v4.x) : cmax;
            cmax = ((word >> (b + 1)) & 1u) ? fmaxf(cmax, v4.y) : cmax;
            cmax = ((word >> (b + 2)) & 1u) ? fmaxf(cmax, v4.z) : cmax;
            cmax = ((word >> (b + 3)) & 1u) ? fmaxf(cmax, v4.w) : cmax;
        }
    }
    cmax = fmaxf(cmax, __shfl_xor(cmax, 16));
    float sc = as_v + cmax;
    float rowmax = (cmax < -1.0e37f) ? 0.f : (sc > 0.f ? sc : LEAKY * sc);

    // Pass 2: streaming P (16x32 bf16) x V (32x128 bf16) with 8 accumulators
    v8f acc[8];
    #pragma unroll
    for (int u = 0; u < 8; ++u)
        #pragma unroll
        for (int j = 0; j < 8; ++j) acc[u][j] = 0.f;
    float rowsum = 0.f;

    for (int t = 0; t < 128; ++t) {                   // m tiles of 32
        int m0 = t * 32;
        uint32_t word = arow[t];
        BF16Frag pa;
        #pragma unroll
        for (int v = 0; v < 8; ++v) {
            int k = kbase(v, half);                   // even -> float2 aligned
            float2 an2 = *(const float2*)(anh + m0 + k);
            float s0 = as_v + an2.x, s1 = as_v + an2.y;
            float l0 = s0 > 0.f ? s0 : LEAKY * s0;
            float l1 = s1 > 0.f ? s1 : LEAKY * s1;
            float p0 = ((word >> k) & 1u)       ? __expf(l0 - rowmax) : 0.f;
            float p1 = ((word >> (k + 1)) & 1u) ? __expf(l1 - rowmax) : 0.f;
            rowsum += p0 + p1;
            pa.u[v] = pack_bf2(p0, p1);
        }
        #pragma unroll
        for (int ut = 0; ut < 8; ++ut) {
            int ucol = ut * 16 + l15;
            BF16Frag pb;
            #pragma unroll
            for (int v = 0; v < 8; ++v) {
                int k = kbase(v, half);
                pb.u[v] = ft32[((h * UU + ucol) * NN + m0 + k) >> 1];
            }
            acc[ut] = __builtin_amdgcn_wmma_f32_16x16x32_bf16(false, pa.v, false, pb.v,
                                                              (short)0, acc[ut], false, false);
        }
    }

    rowsum += __shfl_xor(rowsum, 16);
    float inv = 1.f / fmaxf(rowsum, 1e-20f);
    float invr[8];
    #pragma unroll
    for (int r = 0; r < 8; ++r) invr[r] = __shfl(inv, half * 8 + r); // row r+8*half

    #pragma unroll
    for (int ut = 0; ut < 8; ++ut) {
        int ucol = ut * 16 + l15;
        float bias = biases[h * UU + ucol];
        #pragma unroll
        for (int r = 0; r < 8; ++r) {
            float va = acc[ut][r] * invr[r] + bias;
            float e = va > 0.f ? va : (__expf(va) - 1.f);        // ELU
            atomicAdd(&red[(r + half * 8) * UU + ucol], e * 0.125f); // ds_add_f32
        }
    }
    __syncthreads();
    for (int i = tid; i < 16 * UU; i += 256)
        out[(rb * 16 + (i >> 7)) * UU + (i & 127)] = red[i];
}

// ---------------------------------------------------------------------------
extern "C" void kernel_launch(void* const* d_in, const int* in_sizes, int n_in,
                              void* d_out, int out_size, void* d_ws, size_t ws_size,
                              hipStream_t stream) {
    const float* X          = (const float*)d_in[0];
    const int*   A          = (const int*)d_in[1];
    const float* kernels    = (const float*)d_in[2];
    const float* biases     = (const float*)d_in[3];
    const float* attn_self  = (const float*)d_in[4];
    const float* attn_neigh = (const float*)d_in[5];
    float* out = (float*)d_out;

    char* ws = (char*)d_ws;
    uint32_t* Apack   = (uint32_t*)(ws);                       // 2 MB
    uint16_t* Wt      = (uint16_t*)(ws + 2097152);             // 512 KB
    uint32_t* Xbf     = (uint32_t*)(ws + 2621440);             // 2 MB
    float*    feats   = (float*)   (ws + 4718592);             // 16 MB
    uint16_t* featsT  = (uint16_t*)(ws + 21495808);            // 8 MB
    float*    a_s     = (float*)   (ws + 29884416);            // 128 KB
    float*    a_n     = (float*)   (ws + 30015488);            // 128 KB

    pack_adj_kernel<<<(NN * NN) / 256, 256, 0, stream>>>(A, Apack);
    transpose_w_kernel<<<(HH * UU * FF) / 256, 256, 0, stream>>>(kernels, Wt);
    convert_x_kernel<<<(NN * FF / 2) / 256, 256, 0, stream>>>(X, Xbf);
    feats_kernel<<<(HH * (NN / 16) * (UU / 16)) / 8, 256, 0, stream>>>(
        Xbf, (const uint32_t*)Wt, feats, featsT);
    attn_proj_kernel<<<(HH * NN) / 256, 256, 0, stream>>>(feats, attn_self,
                                                          attn_neigh, a_s, a_n);
    gat_kernel<<<NN / 16, 256, 0, stream>>>(Apack, (const uint32_t*)featsT,
                                            a_s, a_n, biases, out);
}
